// MultiHeadAttention_936302870817
// MI455X (gfx1250) — matrix operand order repaired and checked by reading, not compiled, not run on hardware
//
#include <hip/hip_runtime.h>

typedef __attribute__((ext_vector_type(16))) __bf16 v16bf;
typedef __attribute__((ext_vector_type(8)))  float  v8f;
typedef __attribute__((ext_vector_type(8)))  unsigned short us8;
typedef __attribute__((ext_vector_type(16))) unsigned short us16;
typedef __attribute__((ext_vector_type(4)))  unsigned int u32x4;
typedef __attribute__((ext_vector_type(8)))  int i32x8;
typedef __attribute__((ext_vector_type(4)))  int i32x4;

// ---------- helpers ----------

__device__ __forceinline__ unsigned short f2bf(float f) {
    union { float f; unsigned int u; } c;
    c.f = f;
    unsigned int u = c.u;
    u += 0x7FFFu + ((u >> 16) & 1u);   // round-to-nearest-even
    return (unsigned short)(u >> 16);
}

__device__ __forceinline__ v8f zero8() {
    v8f z = {0.f, 0.f, 0.f, 0.f, 0.f, 0.f, 0.f, 0.f};
    return z;
}

// xor-mask lane swap within each 16-lane row via v_permlane16_b32 (no LDS, no waits)
__device__ __forceinline__ float permxor(float x, unsigned s0, unsigned s1) {
    unsigned u = __builtin_bit_cast(unsigned, x);
    unsigned r = __builtin_amdgcn_permlane16(u, u, s0, s1, false, false);
    return __builtin_bit_cast(float, r);
}
// butterfly all-reduce over the 16 lanes of each half (masks 1,2,4,8)
__device__ __forceinline__ float redmax16(float t) {
    t = fmaxf(t, permxor(t, 0x67452301u, 0xEFCDAB89u));  // xor 1
    t = fmaxf(t, permxor(t, 0x54761032u, 0xDCFE98BAu));  // xor 2
    t = fmaxf(t, permxor(t, 0x32107654u, 0xBA98FEDCu));  // xor 4
    t = fmaxf(t, permxor(t, 0xFEDCBA98u, 0x76543210u));  // xor 8
    return t;
}
__device__ __forceinline__ float redsum16(float t) {
    t += permxor(t, 0x67452301u, 0xEFCDAB89u);
    t += permxor(t, 0x54761032u, 0xDCFE98BAu);
    t += permxor(t, 0x32107654u, 0xBA98FEDCu);
    t += permxor(t, 0xFEDCBA98u, 0x76543210u);
    return t;
}

// A-operand fragment (16x32 bf16), ISA 7.12.2:
// lane<16: K={0..7,16..23}; lane>=16: K={8..15,24..31}; row = lane&15.
__device__ __forceinline__ v16bf ldfragA(const unsigned short* base, int ld, int lane) {
    const int half = (lane >> 4) & 1;
    const int r    = lane & 15;
    const unsigned short* p = base + (size_t)r * ld + half * 8;
    us8 c0 = *(const us8*)(p);
    us8 c1 = *(const us8*)(p + 16);
    us16 w;
#pragma unroll
    for (int i = 0; i < 8; i++) { w[i] = c0[i]; w[i + 8] = c1[i]; }
    return __builtin_bit_cast(v16bf, w);
}

// B-operand fragment (32x16 bf16): lane<16 holds K={0..7,16..23}, lane>=16 K={8..15,24..31};
// column n = lane&15; `base` points at row-major [n][ld] storage.
__device__ __forceinline__ v16bf ldfragB(const unsigned short* base, int ld, int lane) {
    const int half = (lane >> 4) & 1;
    const int c    = lane & 15;
    const unsigned short* p = base + (size_t)c * ld + half * 8;
    us8 c0 = *(const us8*)(p);
    us8 c1 = *(const us8*)(p + 16);
    us16 w;
#pragma unroll
    for (int i = 0; i < 8; i++) { w[i] = c0[i]; w[i + 8] = c1[i]; }
    return __builtin_bit_cast(v16bf, w);
}

__device__ __forceinline__ v8f wmma_bf16(v16bf a, v16bf b, v8f c) {
    return __builtin_amdgcn_wmma_f32_16x16x32_bf16(false, a, false, b, (short)0, c,
                                                   false, false);
}

// ---------- Tensor Data Mover: 2-D bf16 tile -> LDS ----------
// Rows of `tile0` elements (tile0*2 bytes each), `tile1` rows, row stride
// `stride0` elements in memory.  D# pad fields add 32B of LDS padding after
// each 128B of row data -> LDS row stride = tile0 + 16 elements (stride 80
// for tile0==64).  ISA 8.3/8.4 field packing.  6-arg builtin form.
__device__ __forceinline__ void tdm_load_2d(unsigned lds_off, const void* gptr,
                                            unsigned dim0, unsigned dim1,
                                            unsigned tile0, unsigned tile1,
                                            unsigned stride0) {
    const uint64_t ga = (uint64_t)(uintptr_t)gptr;
    u32x4 g0;
    g0[0] = 1u;                                            // count=1, user D#
    g0[1] = lds_off;                                       // lds_addr
    g0[2] = (unsigned)ga;                                  // global_addr[31:0]
    g0[3] = ((unsigned)(ga >> 32) & 0x01FFFFFFu) | 0x80000000u;  // addr[56:32] | type=2
    i32x8 g1;
    // data_size=1 (2B) | pad_enable | pad_interval=4 (32 DW) | pad_amount=7 (8 DW)
    g1[0] = (int)(0x00010000u | 0x00100000u | (4u << 22) | (7u << 25));
    g1[1] = (int)(dim0 << 16);                             // tensor_dim0[15:0]
    g1[2] = (int)((dim0 >> 16) | (dim1 << 16));            // dim0 hi | dim1 lo
    g1[3] = (int)((dim1 >> 16) | (tile0 << 16));           // dim1 hi | tile_dim0
    g1[4] = (int)tile1;                                    // tile_dim1 (tile_dim2=0)
    g1[5] = (int)stride0;                                  // tensor_dim0_stride lo
    g1[6] = 0;
    g1[7] = 0;
    i32x4 z4 = {0, 0, 0, 0};
    i32x8 z8 = {0, 0, 0, 0, 0, 0, 0, 0};
    __builtin_amdgcn_tensor_load_to_lds(g0, g1, z4, z4, z8, 0);
}

__device__ __forceinline__ unsigned lds_off_u32(const void* p) {
    return (unsigned)(uintptr_t)p;   // generic LDS ptr: low 32 bits = LDS offset
}

// ---------- fp32 -> bf16 convert ----------

__global__ void mha_cvt_bf16(const float* __restrict__ in,
                             unsigned short* __restrict__ out, int n) {
    int i  = blockIdx.x * blockDim.x + threadIdx.x;
    int st = gridDim.x * blockDim.x;
    for (; i < n; i += st) out[i] = f2bf(in[i]);
}

// ---------- GEMM: out = A[M x 512] * W[512 x 512]^T + bias ----------
// 128x128 tile, TK=64, 8 waves (4M x 2N), staging via TDM.
// MODE 0: fp32 output [row*512+col];  MODE 1: bf16 head-split [b][h][n][64].

template <int MODE>
__global__ __launch_bounds__(256) void mha_gemm_xwT(
    const unsigned short* __restrict__ A, const unsigned short* __restrict__ W,
    const float* __restrict__ bias, void* __restrict__ outp) {
    __shared__ unsigned short sA[128 * 80];   // 20 KB, row stride 80 (64 data + 16 pad)
    __shared__ unsigned short sB[128 * 80];   // 20 KB

    const int tid    = threadIdx.x;
    const int lane   = tid & 31;
    const int half   = (lane >> 4) & 1;
    const int waveId = tid >> 5;
    const int tileM  = blockIdx.x * 128;
    const int tileN  = blockIdx.y * 128;
    const int wm     = (waveId & 3) * 32;
    const int wn     = (waveId >> 2) * 64;

    v8f acc[2][4];
#pragma unroll
    for (int i = 0; i < 2; i++)
#pragma unroll
        for (int j = 0; j < 4; j++) acc[i][j] = zero8();

    for (int kb = 0; kb < 512; kb += 64) {
        __syncthreads();
        if (waveId == 0) {
            tdm_load_2d(lds_off_u32(sA), A + (size_t)tileM * 512 + kb,
                        512, 8192, 64, 128, 512);
            tdm_load_2d(lds_off_u32(sB), W + (size_t)tileN * 512 + kb,
                        512, 512, 64, 128, 512);
            __builtin_amdgcn_s_wait_tensorcnt(0);
        }
        __syncthreads();

#pragma unroll
        for (int kf = 0; kf < 2; kf++) {
            v16bf af[2], bf[4];
#pragma unroll
            for (int i = 0; i < 2; i++)
                af[i] = ldfragA(&sA[(wm + i * 16) * 80 + kf * 32], 80, lane);
#pragma unroll
            for (int j = 0; j < 4; j++)
                bf[j] = ldfragB(&sB[(wn + j * 16) * 80 + kf * 32], 80, lane);
#pragma unroll
            for (int i = 0; i < 2; i++)
#pragma unroll
                for (int j = 0; j < 4; j++)
                    acc[i][j] = wmma_bf16(af[i], bf[j], acc[i][j]);
        }
    }

#pragma unroll
    for (int i = 0; i < 2; i++)
#pragma unroll
        for (int j = 0; j < 4; j++) {
            const int col = tileN + wn + j * 16 + (lane & 15);
            const float bv = bias[col];
#pragma unroll
            for (int v = 0; v < 8; v++) {
                const int row = tileM + wm + i * 16 + v + 8 * half;
                const float val = acc[i][j][v] + bv;
                if (MODE == 0) {
                    ((float*)outp)[(size_t)row * 512 + col] = val;
                } else {
                    const int b = row >> 11, n = row & 2047;
                    const int h = col >> 6,  d = col & 63;
                    ((unsigned short*)outp)[((((size_t)b * 8 + h) * 2048 + n) << 6) + d] =
                        f2bf(val);
                }
            }
        }
}

// ---------- Flash attention over head-split bf16 Q/K/V ----------
// Q,K,V: [b][h][2048][64] bf16.  Output: [b][2048][512] bf16 (heads merged).

__global__ __launch_bounds__(256) void mha_flash_attn(
    const unsigned short* __restrict__ Qb, const unsigned short* __restrict__ Kb,
    const unsigned short* __restrict__ Vb, unsigned short* __restrict__ Ob) {
    __shared__ unsigned short sK[64 * 80];        // keys row-major [key][d] (TDM, padded)
    __shared__ unsigned short sVt[64 * 80];       // V transposed  [d][key]
    __shared__ unsigned short sP[8 * 32 * 80];    // per-wave P scratch

    const int tid    = threadIdx.x;
    const int lane   = tid & 31;
    const int half   = (lane >> 4) & 1;
    const int waveId = tid >> 5;
    const int bh     = blockIdx.x >> 3;   // b*8 + h
    const int qt     = blockIdx.x & 7;
    const int b      = bh >> 3;
    const int h      = bh & 7;
    const size_t hbase = (size_t)bh * 2048 * 64;
    const unsigned short* Q = Qb + hbase;
    const unsigned short* K = Kb + hbase;
    const unsigned short* V = Vb + hbase;
    const int qRow = qt * 256 + waveId * 32;

    // exp(x/sqrt(512)) == exp2(x*C)
    const float C = 1.4426950408889634f * 0.044194173824159216f;

    v16bf qf[2][2];
#pragma unroll
    for (int fm = 0; fm < 2; fm++)
#pragma unroll
        for (int kf = 0; kf < 2; kf++)
            qf[fm][kf] = ldfragA(Q + (size_t)(qRow + fm * 16) * 64 + kf * 32, 64, lane);

    v8f O[2][4];
#pragma unroll
    for (int i = 0; i < 2; i++)
#pragma unroll
        for (int j = 0; j < 4; j++) O[i][j] = zero8();

    float mrun[2][8], lrun[2][8];
#pragma unroll
    for (int i = 0; i < 2; i++)
#pragma unroll
        for (int j = 0; j < 8; j++) { mrun[i][j] = -3.0e38f; lrun[i][j] = 0.f; }

    unsigned short* pw = &sP[waveId * 32 * 80];

    for (int kb = 0; kb < 2048; kb += 64) {
        __syncthreads();
        if (waveId == 0)   // K block via Tensor Data Mover (padded to stride 80)
            tdm_load_2d(lds_off_u32(sK), K + (size_t)kb * 64, 64, 2048, 64, 64, 64);
        {   // V block staged transposed by all threads
            const int r  = tid >> 2;          // key 0..63
            const int cq = (tid & 3) * 16;    // d base
            const unsigned short* gv = V + (size_t)(kb + r) * 64 + cq;
            us8 v0 = *(const us8*)gv;
            us8 v1 = *(const us8*)(gv + 8);
#pragma unroll
            for (int e = 0; e < 8; e++) sVt[(cq + e) * 80 + r]     = v0[e];
#pragma unroll
            for (int e = 0; e < 8; e++) sVt[(cq + 8 + e) * 80 + r] = v1[e];
        }
        if (waveId == 0) __builtin_amdgcn_s_wait_tensorcnt(0);
        __syncthreads();

        // S = Q * K^T
        v8f S[2][4];
#pragma unroll
        for (int nt = 0; nt < 4; nt++) {
            v16bf kb0 = ldfragB(&sK[(nt * 16) * 80], 80, lane);
            v16bf kb1 = ldfragB(&sK[(nt * 16) * 80 + 32], 80, lane);
#pragma unroll
            for (int fm = 0; fm < 2; fm++) {
                v8f a = zero8();
                a = wmma_bf16(qf[fm][0], kb0, a);
                a = wmma_bf16(qf[fm][1], kb1, a);
                S[fm][nt] = a;
            }
        }

        // online softmax per 16-row group (lane reductions via v_permlane16)
#pragma unroll
        for (int fm = 0; fm < 2; fm++) {
            float mx[8];
#pragma unroll
            for (int j = 0; j < 8; j++)
                mx[j] = redmax16(fmaxf(fmaxf(S[fm][0][j], S[fm][1][j]),
                                       fmaxf(S[fm][2][j], S[fm][3][j])));
            float al[8], sm[8];
#pragma unroll
            for (int j = 0; j < 8; j++) {
                const float mn = fmaxf(mrun[fm][j], mx[j]);
                al[j] = exp2f((mrun[fm][j] - mn) * C);
                mrun[fm][j] = mn;
                sm[j] = 0.f;
            }
#pragma unroll
            for (int nt = 0; nt < 4; nt++)
#pragma unroll
                for (int j = 0; j < 8; j++) {
                    const float p = exp2f((S[fm][nt][j] - mrun[fm][j]) * C);
                    S[fm][nt][j] = p;
                    sm[j] += p;
                }
#pragma unroll
            for (int j = 0; j < 8; j++)
                lrun[fm][j] = lrun[fm][j] * al[j] + redsum16(sm[j]);
#pragma unroll
            for (int dt = 0; dt < 4; dt++)
#pragma unroll
                for (int j = 0; j < 8; j++) O[fm][dt][j] *= al[j];

            // spill P (bf16) into per-wave LDS scratch
#pragma unroll
            for (int nt = 0; nt < 4; nt++)
#pragma unroll
                for (int j = 0; j < 8; j++)
                    pw[(fm * 16 + j + 8 * half) * 80 + nt * 16 + (lane & 15)] =
                        f2bf(S[fm][nt][j]);
        }
        asm volatile("s_wait_dscnt 0" ::: "memory");

        // O += P * V
        v16bf pf[2][2];
#pragma unroll
        for (int fm = 0; fm < 2; fm++)
#pragma unroll
            for (int kf = 0; kf < 2; kf++)
                pf[fm][kf] = ldfragA(&pw[(fm * 16) * 80 + kf * 32], 80, lane);
#pragma unroll
        for (int dt = 0; dt < 4; dt++) {
            v16bf vb0 = ldfragB(&sVt[(dt * 16) * 80], 80, lane);
            v16bf vb1 = ldfragB(&sVt[(dt * 16) * 80 + 32], 80, lane);
#pragma unroll
            for (int fm = 0; fm < 2; fm++) {
                O[fm][dt] = wmma_bf16(pf[fm][0], vb0, O[fm][dt]);
                O[fm][dt] = wmma_bf16(pf[fm][1], vb1, O[fm][dt]);
            }
        }
    }

    // normalize, write merged-head bf16 output [b][n][512]
#pragma unroll
    for (int fm = 0; fm < 2; fm++) {
        float rinv[8];
#pragma unroll
        for (int j = 0; j < 8; j++) rinv[j] = 1.f / lrun[fm][j];
#pragma unroll
        for (int dt = 0; dt < 4; dt++)
#pragma unroll
            for (int j = 0; j < 8; j++) {
                const int n   = qRow + fm * 16 + j + 8 * half;
                const int col = h * 64 + dt * 16 + (lane & 15);
                Ob[((size_t)b * 2048 + n) * 512 + col] = f2bf(O[fm][dt][j] * rinv[j]);
            }
    }
}

// ---------- launch ----------

extern "C" void kernel_launch(void* const* d_in, const int* in_sizes, int n_in,
                              void* d_out, int out_size, void* d_ws, size_t ws_size,
                              hipStream_t stream) {
    const float* x  = (const float*)d_in[0];
    const float* Wq = (const float*)d_in[1];
    const float* bq = (const float*)d_in[2];
    const float* Wk = (const float*)d_in[3];
    const float* bk = (const float*)d_in[4];
    const float* Wv = (const float*)d_in[5];
    const float* bv = (const float*)d_in[6];
    const float* Wo = (const float*)d_in[7];
    const float* bo = (const float*)d_in[8];
    float* out = (float*)d_out;

    const int B = 4, N = 2048, DIM = 512;
    const int M = B * N;                       // 8192
    const size_t xe = (size_t)M * DIM;         // 4,194,304 elems
    const size_t we = (size_t)DIM * DIM;       // 262,144 elems

    unsigned short* xb  = (unsigned short*)d_ws;
    unsigned short* wqb = xb + xe;
    unsigned short* wkb = wqb + we;
    unsigned short* wvb = wkb + we;
    unsigned short* wob = wvb + we;
    unsigned short* Qb  = wob + we;            // [b][h][n][64]
    unsigned short* Kb  = Qb + xe;
    unsigned short* Vb  = Kb + xe;
    unsigned short* Ab  = Vb + xe;             // [b][n][512]

    mha_cvt_bf16<<<1024, 256, 0, stream>>>(x,  xb,  (int)xe);
    mha_cvt_bf16<<<256,  256, 0, stream>>>(Wq, wqb, (int)we);
    mha_cvt_bf16<<<256,  256, 0, stream>>>(Wk, wkb, (int)we);
    mha_cvt_bf16<<<256,  256, 0, stream>>>(Wv, wvb, (int)we);
    mha_cvt_bf16<<<256,  256, 0, stream>>>(Wo, wob, (int)we);

    dim3 gg(M / 128, DIM / 128);
    mha_gemm_xwT<1><<<gg, 256, 0, stream>>>(xb, wqb, bq, Qb);
    mha_gemm_xwT<1><<<gg, 256, 0, stream>>>(xb, wkb, bk, Kb);
    mha_gemm_xwT<1><<<gg, 256, 0, stream>>>(xb, wvb, bv, Vb);

    mha_flash_attn<<<B * 8 * (N / 256), 256, 0, stream>>>(Qb, Kb, Vb, Ab);

    mha_gemm_xwT<0><<<gg, 256, 0, stream>>>(Ab, wob, bo, out);
}
